// CorticalColumnModel_30425548324897
// MI455X (gfx1250) — compile-verified, matrix operand order checked by reading, not run on hardware
//
#include <hip/hip_runtime.h>
#include <stdint.h>
#include <stddef.h>

// ---------------------------------------------------------------------------
// Types
// ---------------------------------------------------------------------------
typedef __attribute__((ext_vector_type(16))) __bf16 v16bf;
typedef __attribute__((ext_vector_type(8)))  __bf16 bf16x8;
typedef __attribute__((ext_vector_type(8)))  float  v8f;

union FragU {
    v16bf v;
    struct { bf16x8 lo; bf16x8 hi; } s;
};

__device__ __forceinline__ __bf16 f2bf(float f) {
    unsigned int u = __builtin_bit_cast(unsigned int, f);
    unsigned int r = u + 0x7FFFu + ((u >> 16) & 1u);   // round-to-nearest-even
    unsigned short h = (unsigned short)(r >> 16);
    return __builtin_bit_cast(__bf16, h);
}
__device__ __forceinline__ float bf2f(__bf16 b) {
    unsigned short h = __builtin_bit_cast(unsigned short, b);
    unsigned int u = ((unsigned int)h) << 16;
    return __builtin_bit_cast(float, u);
}

// CDNA5 async global->LDS copy (16B per lane), tracked on ASYNCcnt.
__device__ __forceinline__ void async_copy_b128(uint32_t ldsOff, const void* gaddr) {
    asm volatile("global_load_async_to_lds_b128 %0, %1, off"
                 :: "v"(ldsOff), "v"((uint64_t)(uintptr_t)gaddr)
                 : "memory");
}
__device__ __forceinline__ void wait_async0() {
    asm volatile("s_wait_asynccnt 0x0" ::: "memory");
}

// ---------------------------------------------------------------------------
// Model constants
// ---------------------------------------------------------------------------
static const int Bq   = 16384;   // batch
static const int Dq   = 1024;    // width
static const int Lq   = 6;       // layers
static const int NMCq = 64;
static const int Hq   = 16;

// GEMM tiling
#define BM  128
#define BN  128
#define BKS 64          // K elements staged per barrier (two wmma K-steps)

// Epilogue modes
#define MODE_BIAS      0   // y = acc + b
#define MODE_BIAS_RELU 1   // y = relu(acc + b)
#define MODE_SUB_RELU  2   // y = relu(aux_bf16 - (acc + b))
#define MODE_ADD_F32   3   // y = aux_f32 + acc + b

// ---------------------------------------------------------------------------
// Main WMMA GEMM: C[M,N] = A[M,K] (bf16, row-major) * Wt[N,K] (bf16, i.e. W^T)
// 256 threads = 8 waves; block tile 128x128; wave tile 32x64 (2x4 wmma tiles).
// Double-buffered LDS fed by global_load_async_to_lds_b128; one barrier/stage.
// ---------------------------------------------------------------------------
__global__ __launch_bounds__(256)
void k_gemm_bf16_wmma(const __bf16* __restrict__ A,
                      const __bf16* __restrict__ Wt,
                      const float*  __restrict__ bias,
                      const __bf16* __restrict__ auxB,
                      const float*  __restrict__ auxF,
                      float*        __restrict__ outF,
                      __bf16*       __restrict__ outB,
                      int M, int N, int K, int mode)
{
    __shared__ __align__(16) __bf16 sA[2][BM * BKS];   // 2 x 16 KB
    __shared__ __align__(16) __bf16 sB[2][BN * BKS];   // 2 x 16 KB

    const int tid  = threadIdx.x;
    const int wave = tid >> 5;            // 0..7 (wave32)
    const int lane = tid & 31;
    const int wm   = wave >> 1;           // 0..3 : M direction
    const int wn   = wave & 1;            // 0..1 : N direction
    const int rl   = lane & 15;
    const int hs   = lane >> 4;           // lane half-select
    const int kh   = hs * 8;              // K sub-offset per ISA 16-bit layout

    const int bM = blockIdx.y * BM;
    const int bN = blockIdx.x * BN;

    // per-thread staging chunk coordinates: 1024 x 16B chunks per tile, 4/thread
    int crow[4], ccol[4];
#pragma unroll
    for (int t = 0; t < 4; ++t) {
        int chunk = tid + t * 256;
        crow[t] = chunk >> 3;             // 0..127
        ccol[t] = (chunk & 7) * 8;        // element offset within row (0..56)
    }

    v8f acc[2][4];
#pragma unroll
    for (int mt = 0; mt < 2; ++mt)
#pragma unroll
        for (int nt = 0; nt < 4; ++nt)
            acc[mt][nt] = (v8f)(0.0f);

    const int nstages = K / BKS;

    // ---- prologue: stage 0 into buffer 0 ----
    {
        uint32_t aBase = (uint32_t)(uintptr_t)(&sA[0][0]);
        uint32_t bBase = (uint32_t)(uintptr_t)(&sB[0][0]);
#pragma unroll
        for (int t = 0; t < 4; ++t) {
            uint32_t lo = (uint32_t)(crow[t] * BKS + ccol[t]) * 2u;
            async_copy_b128(aBase + lo, A  + (size_t)(bM + crow[t]) * K + ccol[t]);
            async_copy_b128(bBase + lo, Wt + (size_t)(bN + crow[t]) * K + ccol[t]);
        }
    }

    for (int s = 0; s < nstages; ++s) {
        wait_async0();        // my async writes for stage s landed
        __syncthreads();      // everyone's landed; other buffer free to overwrite

        // ---- issue async copy of stage s+1 (overlaps with wmma below) ----
        if (s + 1 < nstages) {
            const int nb = (s + 1) & 1;
            const int k1 = (s + 1) * BKS;
            uint32_t aBase = (uint32_t)(uintptr_t)(&sA[nb][0]);
            uint32_t bBase = (uint32_t)(uintptr_t)(&sB[nb][0]);
#pragma unroll
            for (int t = 0; t < 4; ++t) {
                uint32_t lo = (uint32_t)(crow[t] * BKS + ccol[t]) * 2u;
                async_copy_b128(aBase + lo, A  + (size_t)(bM + crow[t]) * K + k1 + ccol[t]);
                async_copy_b128(bBase + lo, Wt + (size_t)(bN + crow[t]) * K + k1 + ccol[t]);
            }
            // light prefetch of stage s+2 into L2 (global_prefetch_b8)
            if (s + 2 < nstages) {
                __builtin_prefetch(A  + (size_t)(bM + (tid >> 1)) * K + k1 + BKS + (tid & 1) * 32, 0, 1);
                __builtin_prefetch(Wt + (size_t)(bN + (tid >> 1)) * K + k1 + BKS + (tid & 1) * 32, 0, 1);
            }
        }

        // ---- compute stage s: two 32-wide K-steps, 16 wmma ----
        const __bf16* cA = &sA[s & 1][0];
        const __bf16* cB = &sB[s & 1][0];
#pragma unroll
        for (int ks = 0; ks < 2; ++ks) {
            const int ko = ks * 32;
            FragU fa[2], fb[4];
#pragma unroll
            for (int mt = 0; mt < 2; ++mt) {
                int r = wm * 32 + mt * 16 + rl;
                fa[mt].s.lo = *reinterpret_cast<const bf16x8*>(&cA[r * BKS + ko + kh]);
                fa[mt].s.hi = *reinterpret_cast<const bf16x8*>(&cA[r * BKS + ko + 16 + kh]);
            }
#pragma unroll
            for (int nt = 0; nt < 4; ++nt) {
                int r = wn * 64 + nt * 16 + rl;
                fb[nt].s.lo = *reinterpret_cast<const bf16x8*>(&cB[r * BKS + ko + kh]);
                fb[nt].s.hi = *reinterpret_cast<const bf16x8*>(&cB[r * BKS + ko + 16 + kh]);
            }
#pragma unroll
            for (int mt = 0; mt < 2; ++mt)
#pragma unroll
                for (int nt = 0; nt < 4; ++nt)
                    acc[mt][nt] = __builtin_amdgcn_wmma_f32_16x16x32_bf16(
                        false, fa[mt].v, false, fb[nt].v,
                        (short)0, acc[mt][nt], false, false);
        }
    }

    // ---- epilogue: C layout per ISA (VGPR r: M = r + hs*8, N = rl) ----
#pragma unroll
    for (int mt = 0; mt < 2; ++mt) {
#pragma unroll
        for (int nt = 0; nt < 4; ++nt) {
            int baseR = bM + wm * 32 + mt * 16 + hs * 8;
            int baseC = bN + wn * 64 + nt * 16 + rl;
            float bc = bias ? bias[baseC] : 0.0f;
#pragma unroll
            for (int r = 0; r < 8; ++r) {
                size_t idx = (size_t)(baseR + r) * N + baseC;
                float v = acc[mt][nt][r] + bc;
                if (mode == MODE_BIAS_RELU)      v = fmaxf(v, 0.0f);
                else if (mode == MODE_SUB_RELU)  v = fmaxf(bf2f(auxB[idx]) - v, 0.0f);
                else if (mode == MODE_ADD_F32)   v = auxF[idx] + v;
                if (outF) outF[idx] = v;
                if (outB) outB[idx] = f2bf(v);
            }
        }
    }
}

// ---------------------------------------------------------------------------
// Conversion / repacking kernels
// ---------------------------------------------------------------------------
__global__ void k_cvt_f32_bf16(const float* __restrict__ in, __bf16* __restrict__ out, size_t n) {
    size_t i = (size_t)blockIdx.x * blockDim.x + threadIdx.x;
    if (i < n) out[i] = f2bf(in[i]);
}

// We/Wi [L,NMC,D,H] f32 -> Wt [L, Dout, Din] bf16 : Wt[l][o][d] = W[l][o>>4][d][o&15]
__global__ void k_repack_mc(const float* __restrict__ in, __bf16* __restrict__ out, size_t n) {
    size_t i = (size_t)blockIdx.x * blockDim.x + threadIdx.x;
    if (i >= n) return;
    size_t l   = i >> 20;                  // D*D == 1<<20
    size_t rem = i & ((1u << 20) - 1u);
    size_t o   = rem >> 10;
    size_t d   = rem & 1023u;
    size_t src = (l << 20) + (o >> 4) * (size_t)(Dq * Hq) + d * Hq + (o & 15u);
    out[i] = f2bf(in[src]);
}

// Wl [L,NMC,H,H] -> transposed block-diagonal [L, Dout, Din] bf16
__global__ void k_repack_wl(const float* __restrict__ in, __bf16* __restrict__ out, size_t n) {
    size_t i = (size_t)blockIdx.x * blockDim.x + threadIdx.x;
    if (i >= n) return;
    size_t l   = i >> 20;
    size_t rem = i & ((1u << 20) - 1u);
    size_t o   = rem >> 10;
    size_t d   = rem & 1023u;
    float v = 0.0f;
    if ((o >> 4) == (d >> 4)) {
        size_t src = l * (size_t)(NMCq * Hq * Hq) + (o >> 4) * 256u + (d & 15u) * 16u + (o & 15u);
        v = in[src];
    }
    out[i] = f2bf(v);
}

// Dense [L,D,D] f32 -> transposed bf16 [L,D,D]: out[l][o][d] = in[l][d][o]
__global__ __launch_bounds__(256)
void k_transpose_cvt(const float* __restrict__ in, __bf16* __restrict__ out, int Dd) {
    __shared__ float t[32][33];
    const float* src = in  + (size_t)blockIdx.z * Dd * Dd;
    __bf16*      dst = out + (size_t)blockIdx.z * Dd * Dd;
    int x = blockIdx.x * 32 + threadIdx.x;
    int y = blockIdx.y * 32 + threadIdx.y;
#pragma unroll
    for (int j = 0; j < 32; j += 8)
        t[threadIdx.y + j][threadIdx.x] = src[(size_t)(y + j) * Dd + x];
    __syncthreads();
    int x2 = blockIdx.y * 32 + threadIdx.x;
    int y2 = blockIdx.x * 32 + threadIdx.y;
#pragma unroll
    for (int j = 0; j < 32; j += 8)
        dst[(size_t)(y2 + j) * Dd + x2] = f2bf(t[threadIdx.x][threadIdx.y + j]);
}

// ---------------------------------------------------------------------------
// Host
// ---------------------------------------------------------------------------
static inline char* ws_align(char*& p, size_t bytes) {
    uintptr_t v = (uintptr_t)p;
    v = (v + 255u) & ~(uintptr_t)255u;
    char* r = (char*)v;
    p = r + bytes;
    return r;
}

extern "C" void kernel_launch(void* const* d_in, const int* in_sizes, int n_in,
                              void* d_out, int out_size, void* d_ws, size_t ws_size,
                              hipStream_t stream) {
    (void)in_sizes; (void)n_in; (void)out_size; (void)ws_size;

    const float* x    = (const float*)d_in[0];
    const float* We   = (const float*)d_in[1];
    const float* be   = (const float*)d_in[2];
    const float* Wi   = (const float*)d_in[3];
    const float* bi   = (const float*)d_in[4];
    const float* Wl   = (const float*)d_in[5];
    const float* bl   = (const float*)d_in[6];
    const float* Wlat = (const float*)d_in[7];
    const float* blat = (const float*)d_in[8];
    const float* Wv   = (const float*)d_in[9];
    const float* bv   = (const float*)d_in[10];
    const float* Wo   = (const float*)d_in[11];
    const float* bo   = (const float*)d_in[12];
    const float* fbW  = (const float*)d_in[13];
    const float* fbb  = (const float*)d_in[14];

    float* out = (float*)d_out;

    const size_t BD = (size_t)Bq * Dq;       // activation elements
    const size_t DD = (size_t)Dq * Dq;       // one weight matrix

    // ---- workspace carve-out ----
    char* p = (char*)d_ws;
    __bf16* actX = (__bf16*)ws_align(p, BD * 2);
    __bf16* T0   = (__bf16*)ws_align(p, BD * 2);
    __bf16* T1   = (__bf16*)ws_align(p, BD * 2);
    __bf16* T2   = (__bf16*)ws_align(p, BD * 2);
    __bf16* WeT   = (__bf16*)ws_align(p, (size_t)Lq * DD * 2);
    __bf16* WiT   = (__bf16*)ws_align(p, (size_t)Lq * DD * 2);
    __bf16* WlT   = (__bf16*)ws_align(p, (size_t)Lq * DD * 2);
    __bf16* WlatT = (__bf16*)ws_align(p, (size_t)Lq * DD * 2);
    __bf16* WvT   = (__bf16*)ws_align(p, (size_t)Lq * DD * 2);
    __bf16* WoT   = (__bf16*)ws_align(p, (size_t)Lq * DD * 2);
    __bf16* fbWT  = (__bf16*)ws_align(p, (size_t)(Lq - 1) * DD * 2);

    // ---- conversions / repacks ----
    {
        size_t n = BD;
        k_cvt_f32_bf16<<<(unsigned)((n + 255) / 256), 256, 0, stream>>>(x, actX, n);
        size_t nw = (size_t)Lq * DD;
        k_repack_mc<<<(unsigned)((nw + 255) / 256), 256, 0, stream>>>(We, WeT, nw);
        k_repack_mc<<<(unsigned)((nw + 255) / 256), 256, 0, stream>>>(Wi, WiT, nw);
        k_repack_wl<<<(unsigned)((nw + 255) / 256), 256, 0, stream>>>(Wl, WlT, nw);
        dim3 tg(Dq / 32, Dq / 32, Lq), tb(32, 8);
        k_transpose_cvt<<<tg, tb, 0, stream>>>(Wlat, WlatT, Dq);
        k_transpose_cvt<<<tg, tb, 0, stream>>>(Wv, WvT, Dq);
        k_transpose_cvt<<<tg, tb, 0, stream>>>(Wo, WoT, Dq);
        dim3 tgf(Dq / 32, Dq / 32, Lq - 1);
        k_transpose_cvt<<<tgf, tb, 0, stream>>>(fbW, fbWT, Dq);
    }

    const dim3 gg(Dq / BN, Bq / BM);   // (8, 128)
    const dim3 gb(256);

#define GEMM(Ain, Wm, bvec, auxb, auxf, of, ob, md) \
    k_gemm_bf16_wmma<<<gg, gb, 0, stream>>>((Ain), (Wm), (bvec), (auxb), (auxf), (of), (ob), Bq, Dq, Dq, (md))

    // ---- forward layers ----
    __bf16* cur = actX;
    for (int l = 0; l < Lq; ++l) {
        const __bf16* we = WeT + (size_t)l * DD;
        const __bf16* wi = WiT + (size_t)l * DD;
        const __bf16* wl = WlT + (size_t)l * DD;
        const __bf16* wlat = WlatT + (size_t)l * DD;
        const __bf16* wv = WvT + (size_t)l * DD;
        const __bf16* wo = WoT + (size_t)l * DD;
        const float* be_l = be + (size_t)l * Dq;
        const float* bi_l = bi + (size_t)l * Dq;
        const float* bl_l = bl + (size_t)l * Dq;
        const float* blat_l = blat + (size_t)l * Dq;
        const float* bv_l = bv + (size_t)l * Dq;
        const float* bo_l = bo + (size_t)l * Dq;
        float* out_l = out + (size_t)l * BD;

        GEMM(cur, we,   be_l,   nullptr, nullptr, nullptr, T0, MODE_BIAS_RELU);  // exc
        GEMM(cur, wi,   bi_l,   nullptr, nullptr, nullptr, T1, MODE_BIAS_RELU);  // inh
        GEMM(T1,  wl,   bl_l,   T0,      nullptr, nullptr, T2, MODE_SUB_RELU);   // mc = relu(exc-lat)
        GEMM(T2,  wlat, blat_l, nullptr, nullptr, nullptr, T0, MODE_BIAS);       // lateral
        GEMM(T0,  wv,   bv_l,   nullptr, nullptr, nullptr, T1, MODE_BIAS);       // v_proj
        GEMM(T1,  wo,   bo_l,   nullptr, nullptr, out_l,   T2, MODE_BIAS);       // out_proj
        cur = T2;                                                                // next layer input
    }

    // ---- feedback: outs[idx] += fb_in @ fbW[i] + fbb[i]; fb_in = outs[idx] ----
    __bf16* fbA = cur;   // bf16 copy of outs[L-1]
    for (int i = 0; i < Lq - 1; ++i) {
        int idx = Lq - 2 - i;
        float* slice = out + (size_t)idx * BD;
        __bf16* nxt = (fbA == T0) ? T1 : T0;
        GEMM(fbA, fbWT + (size_t)i * DD, fbb + (size_t)i * Dq,
             nullptr, slice, slice, nxt, MODE_ADD_F32);
        fbA = nxt;
    }
#undef GEMM
}
